// DiagonalLSTM_67053029425630
// MI455X (gfx1250) — compile-verified, matrix operand order checked by reading, not run on hardware
//
#include <hip/hip_runtime.h>
#include <hip/hip_bf16.h>
#include <stdint.h>

// ---------------------------------------------------------------------------
// DiagonalLSTM on MI455X (gfx1250, wave32, WMMA).
//   phase 0 (prep): bf16-pack weights, fold biases.
//   phase 1 (feat): skew + 1x1-conv GEMM via v_wmma_f32_16x16x32_bf16,
//                   results stored in WMMA *fragment order* (bf16) so the
//                   consumer re-loads each tile with one b128 per lane.
//   phase 2 (lstm): 63-step diagonal recurrence; one block per batch;
//                   Wcat(512x256 bf16) + h-state in LDS (278KB of 320KB WGP),
//                   f32 cell state in VGPRs, i/g/f/o tiles per-wave so the
//                   nonlinear update is fully in-register; feat fragments
//                   software-pipelined across steps; weights staged to LDS
//                   with GLOBAL_LOAD_ASYNC_TO_LDS_B128 (ASYNCcnt).
// ---------------------------------------------------------------------------

#define SEQ 63  // 2*32 - 1 diagonals

typedef __attribute__((ext_vector_type(16))) __bf16 v16bf;
typedef __attribute__((ext_vector_type(8)))  __bf16 bf16x8;
typedef __attribute__((ext_vector_type(8)))  float  v8f;

union ABFrag { v16bf v; bf16x8 h[2]; };
union Pack8  { bf16x8 v; uint4 u; };

static __device__ __forceinline__ __bf16 tobf(float x) { return (__bf16)x; }
static __device__ __forceinline__ float sigf(float x) { return 1.0f / (1.0f + __expf(-x)); }
static __device__ __forceinline__ float tanh_fast(float x) {
    float e = __expf(2.0f * x);
    return (e - 1.0f) / (e + 1.0f);
}

// CDNA5 async Global->LDS copy, 16 bytes per lane, tracked by ASYNCcnt.
// LDS flat addresses truncate to the 32-bit LDS offset (ISA 10.2).
static __device__ __forceinline__ void async_copy_b128(void* lds_ptr, const void* gptr) {
#if defined(__gfx1250__)
    uint32_t lds_off = (uint32_t)(uintptr_t)lds_ptr;
    asm volatile("global_load_async_to_lds_b128 %0, %1, off"
                 :: "v"(lds_off), "v"(gptr) : "memory");
#endif
}
static __device__ __forceinline__ void async_wait_all() {
#if defined(__gfx1250__)
    asm volatile("s_wait_asynccnt 0x0" ::: "memory");
#endif
}

// 16x32 bf16 fragment (A layout / transposed-B layout) from a row-major
// buffer: lane (m = lane&15, hk = lane>>4) does two contiguous b128 loads.
static __device__ __forceinline__ v16bf load_frag(const __bf16* tile, int rowstride,
                                                  int m, int hk) {
    ABFrag f;
    const __bf16* p = tile + m * rowstride + hk * 8;
    f.h[0] = *(const bf16x8*)(p);
    f.h[1] = *(const bf16x8*)(p + 16);
    return f.v;
}

// ---------------------------------------------------------------------------
// Phase 0: pack weights.
//   wsWcat[o][cc]: cc<128 -> stos_w[o][cc][0] (multiplies shifted h)
//                  cc>=128 -> stos_w[o][cc-128][1] (multiplies h)
//   wsWi  [o][c] : itos_w bf16
//   wsBias[o]    : itos_b + stos_b (folded into feat)
// ---------------------------------------------------------------------------
__global__ __launch_bounds__(256) void prep_kernel(
    const float* __restrict__ itos_w, const float* __restrict__ itos_b,
    const float* __restrict__ stos_w, const float* __restrict__ stos_b,
    __bf16* __restrict__ wsWcat, __bf16* __restrict__ wsWi,
    float* __restrict__ wsBias)
{
    int tid = blockIdx.x * 256 + threadIdx.x;   // 0 .. 131071
    {
        int o = tid >> 8, cc = tid & 255;
        int c = cc & 127, k = cc >> 7;
        wsWcat[tid] = tobf(stos_w[(o * 128 + c) * 2 + k]);
    }
    if (tid < 65536) wsWi[tid] = tobf(itos_w[tid]);
    if (tid < 512)   wsBias[tid] = itos_b[tid] + stos_b[tid];
}

// ---------------------------------------------------------------------------
// Phase 1: feat[b, n, o, s] = sum_c itos_w[o,c]*image[b,c,s,n-s] (+bias fold)
// One block per (n, b).  8 waves, each wave: 4 o-tiles x 2 s-tiles, K=128.
// Output in fragment order: feat[b][n][ot(32)][st(2)][lane(32)][v(8)] bf16.
// ---------------------------------------------------------------------------
__global__ __launch_bounds__(256) void feat_kernel(
    const float* __restrict__ image, const __bf16* __restrict__ wsWi,
    const float* __restrict__ wsBias, __bf16* __restrict__ wsFeat)
{
    extern __shared__ char smem[];
    __bf16* Wi = (__bf16*)smem;                // 512 x 128 bf16   (128 KB)
    __bf16* Xt = (__bf16*)(smem + 131072);     //  32 x 128 bf16 [s][c] (8 KB)
    const int n = blockIdx.x, b = blockIdx.y;
    const int tid = threadIdx.x;

    // stage weights into LDS via async DMA (8192 x 16B, 32 per thread)
    for (int i = tid; i < 8192; i += 256)
        async_copy_b128((uint4*)Wi + i, (const uint4*)wsWi + i);

    // gather skewed column, transposed for B fragments
    for (int i = tid; i < 4096; i += 256) {
        int c = i >> 5, ss = i & 31;
        int col = n - ss;
        float v = (col >= 0 && col < 32)
                    ? image[(((size_t)b * 128 + c) * 32 + ss) * 32 + col] : 0.0f;
        Xt[ss * 128 + c] = tobf(v);
    }
    async_wait_all();
    __syncthreads();

    const int lane = tid & 31, w = tid >> 5;
    const int m = lane & 15, hk = lane >> 4;
    __bf16* featb = wsFeat + (size_t)(b * SEQ + n) * (32 * 2 * 256);

    for (int t = 0; t < 4; ++t) {
        int ot = w * 4 + t;
        int o0 = ot * 16;
        for (int st = 0; st < 2; ++st) {
            int s0 = st * 16;
            v8f acc;
            #pragma unroll
            for (int v = 0; v < 8; ++v) acc[v] = wsBias[o0 + v + 8 * hk];
            #pragma unroll
            for (int kt = 0; kt < 4; ++kt) {
                v16bf a  = load_frag(Wi + o0 * 128 + kt * 32, 128, m, hk);
                v16bf bb = load_frag(Xt + s0 * 128 + kt * 32, 128, m, hk);
                acc = __builtin_amdgcn_wmma_f32_16x16x32_bf16(
                        false, a, false, bb, (short)0, acc, false, false);
            }
            // packed fragment-order store: one b128 per lane
            Pack8 pk;
            #pragma unroll
            for (int v = 0; v < 8; ++v) pk.v[v] = tobf(acc[v]);
            *(uint4*)(featb + ((size_t)(ot * 2 + st) * 32 + lane) * 8) = pk.u;
        }
    }
}

// ---------------------------------------------------------------------------
// Phase 2: the recurrence.  One block per batch, 16 waves.
// Wave w owns channel tile r=(w>>1)*16, column tile s0=(w&1)*16, and computes
// its own i/g/f/o gate tiles (o-tiles g*8 + r/16) so the cell update is
// register-local; f32 cell state lives in 8 VGPRs per lane for all 63 steps.
// ---------------------------------------------------------------------------
__global__ __launch_bounds__(512) void lstm_kernel(
    const __bf16* __restrict__ wsWcat, const __bf16* __restrict__ wsFeat,
    const float* __restrict__ h0, const float* __restrict__ c0,
    float* __restrict__ out)
{
    extern __shared__ char smem[];
    __bf16* Wc   = (__bf16*)smem;              // 512 x 256 bf16 (256 KB)
    __bf16* hcat = (__bf16*)(smem + 262144);   //  32 x 256 bf16 [s][cc] (16 KB)
    const int b = blockIdx.x, tid = threadIdx.x;

    // stage concatenated recurrent weights into LDS via async DMA
    for (int i = tid; i < 16384; i += 512)
        async_copy_b128((uint4*)Wc + i, (const uint4*)wsWcat + i);

    // init h state: cc<128 = shifted (h[.,s-1], row 0 stays zero forever),
    //               cc>=128 = current h
    for (int i = tid; i < 8192; i += 512) {
        int ss = i >> 8, cc = i & 255;
        float v;
        if (cc < 128) v = (ss > 0) ? h0[cc * 32 + ss - 1] : 0.0f;
        else          v = h0[(cc - 128) * 32 + ss];
        hcat[i] = tobf(v);
    }

    const int lane = tid & 31, w = tid >> 5;
    const int nn = lane & 15, hk = lane >> 4;
    const int rt = w >> 1, stI = w & 1;
    const int r = rt * 16, s0 = stI * 16;
    const int s = s0 + nn;                      // diagonal row owned by this lane col

    float creg[8];
    #pragma unroll
    for (int v = 0; v < 8; ++v) creg[v] = c0[(r + v + 8 * hk) * 32 + s];

    const __bf16* fbase = wsFeat + (size_t)b * SEQ * (32 * 2 * 256);
    size_t foff[4];
    #pragma unroll
    for (int g = 0; g < 4; ++g)
        foff[g] = ((size_t)((g * 8 + rt) * 2 + stI) * 32 + lane) * 8;

    // software pipeline: fragments for step 0 loaded up-front
    bf16x8 fcur[4];
    #pragma unroll
    for (int g = 0; g < 4; ++g) fcur[g] = *(const bf16x8*)(fbase + foff[g]);

    async_wait_all();
    __syncthreads();

    for (int n = 0; n < SEQ; ++n) {
        // B fragments: whole K=256 state column for this s-tile (8 k-tiles)
        ABFrag bfr[8];
        const __bf16* hrow = hcat + s * 256;
        #pragma unroll
        for (int kt = 0; kt < 8; ++kt) {
            const __bf16* p = hrow + kt * 32 + hk * 8;
            bfr[kt].h[0] = *(const bf16x8*)(p);
            bfr[kt].h[1] = *(const bf16x8*)(p + 16);
        }

        // accumulators init from this step's (already resident) fragments
        v8f acc[4];
        #pragma unroll
        for (int g = 0; g < 4; ++g) {
            #pragma unroll
            for (int v = 0; v < 8; ++v) acc[g][v] = (float)fcur[g][v];
        }

        // prefetch next step's feat fragments; overlaps the WMMA chain below
        const __bf16* fnx = fbase + (size_t)((n + 1 < SEQ) ? n + 1 : n) * (32 * 2 * 256);
        bf16x8 fnext[4];
        #pragma unroll
        for (int g = 0; g < 4; ++g) fnext[g] = *(const bf16x8*)(fnx + foff[g]);

        #pragma unroll
        for (int g = 0; g < 4; ++g) {
            int row0 = g * 128 + r;
            #pragma unroll
            for (int kt = 0; kt < 8; ++kt) {
                v16bf a = load_frag(Wc + row0 * 256 + kt * 32, 256, nn, hk);
                acc[g] = __builtin_amdgcn_wmma_f32_16x16x32_bf16(
                            false, a, false, bfr[kt].v, (short)0, acc[g], false, false);
            }
        }

        // cell update (matching i/g/f/o lane layout -> pure register math)
        float hn[8];
        #pragma unroll
        for (int v = 0; v < 8; ++v) {
            float i_ = acc[0][v], g_ = acc[1][v], f_ = acc[2][v], o_ = acc[3][v];
            float cn = sigf(f_) * creg[v] + sigf(i_) * tanh_fast(g_);
            hn[v]    = creg[v] * sigf(o_);      // reference uses c_prev here
            creg[v]  = cn;
        }

        __syncthreads();    // all waves done reading hcat for this step

        // packed h writeback: this lane's 8 channels are contiguous
        Pack8 hu;
        #pragma unroll
        for (int v = 0; v < 8; ++v) hu.v[v] = tobf(hn[v]);
        const int chbase = r + 8 * hk;
        *(uint4*)&hcat[s * 256 + 128 + chbase] = hu.u;              // current
        if (s < 31) *(uint4*)&hcat[(s + 1) * 256 + chbase] = hu.u;  // shifted

        int j = n - s;      // unskewed output column
        if (j >= 0 && j < 32) {
            #pragma unroll
            for (int v = 0; v < 8; ++v)
                out[(((size_t)b * 128 + chbase + v) * 32 + s) * 32 + j] = hn[v];
        }
        __syncthreads();    // writes visible before next step's loads

        #pragma unroll
        for (int g = 0; g < 4; ++g) fcur[g] = fnext[g];
    }
}

// ---------------------------------------------------------------------------
extern "C" void kernel_launch(void* const* d_in, const int* in_sizes, int n_in,
                              void* d_out, int out_size, void* d_ws, size_t ws_size,
                              hipStream_t stream) {
    const float* image  = (const float*)d_in[0];
    const float* itos_w = (const float*)d_in[1];
    const float* itos_b = (const float*)d_in[2];
    const float* stos_w = (const float*)d_in[3];
    const float* stos_b = (const float*)d_in[4];
    const float* h0     = (const float*)d_in[5];
    const float* c0     = (const float*)d_in[6];
    float* out = (float*)d_out;

    char* ws = (char*)d_ws;
    __bf16* wsWcat = (__bf16*)(ws);                            // 262144 B
    __bf16* wsWi   = (__bf16*)(ws + 262144);                   // 131072 B
    float*  wsBias = (float*) (ws + 262144 + 131072);          //   2048 B
    __bf16* wsFeat = (__bf16*)(ws + 262144 + 131072 + 4096);   // 128*63*16384*2 B

    prep_kernel<<<512, 256, 0, stream>>>(itos_w, itos_b, stos_w, stos_b,
                                         wsWcat, wsWi, wsBias);
    feat_kernel<<<dim3(SEQ, 128), 256, 131072 + 8192, stream>>>(
        image, wsWi, wsBias, wsFeat);
    lstm_kernel<<<128, 512, 262144 + 16384, stream>>>(
        wsWcat, wsFeat, h0, c0, out);
}
